// MoeTransformerBlock_2551210573880
// MI455X (gfx1250) — compile-verified, matrix operand order checked by reading, not run on hardware
//
#include <hip/hip_runtime.h>

// ---------------------------------------------------------------------------
// MoE transformer block for gfx1250 (MI455X).
// bf16 WMMA GEMMs fed by the Tensor Data Mover (double-buffered LDS),
// flash-style attention with exact softmax + head-averaged pattern output.
// B=2, S=2048, D=1024, H=16, DH=64, E=8, top-K=2, DFF=4096.
// ---------------------------------------------------------------------------

#define DEV __device__ __forceinline__

typedef __attribute__((ext_vector_type(16))) __bf16 v16bf;
typedef __attribute__((ext_vector_type(8)))  float  v8f;
typedef __attribute__((ext_vector_type(4)))  unsigned int u32x4;
typedef __attribute__((ext_vector_type(8)))  int          i32x8;
typedef __attribute__((ext_vector_type(4)))  int          i32x4;

union FragU { v16bf v; unsigned int u[8]; };

DEV unsigned short f2bf(float x) {            // round-to-nearest-even f32->bf16
  unsigned int u = __float_as_uint(x);
  u += 0x7FFFu + ((u >> 16) & 1u);
  return (unsigned short)(u >> 16);
}

DEV v8f wmma_bf16(v16bf a, v16bf b, v8f c) {
  // D = A(16x32 bf16) x B(32x16 bf16) + C(16x16 f32)
  return __builtin_amdgcn_wmma_f32_16x16x32_bf16(
      /*neg_a=*/false, a, /*neg_b=*/false, b,
      /*c_mod=*/(short)0, c, /*reuse_a=*/false, /*reuse_b=*/false);
}

// Load a 16x32 bf16 A-fragment (or B-fragment of W^T) from a u32 view.
// ISA 7.12.2 (16-bit A 16x32): lane m=lane&15, k-pair index
//   kp(v) = (v<4 ? v : v+4) + 4*(lane>>4)   (u32 = 2 packed bf16)
DEV v16bf load_frag(const unsigned int* base, int row0, int rstride, int cu0) {
  int lane = threadIdx.x & 31;
  const unsigned int* p = base + (row0 + (lane & 15)) * rstride + cu0 + ((lane >> 4) << 2);
  FragU f;
  f.u[0] = p[0]; f.u[1] = p[1]; f.u[2] = p[2];  f.u[3] = p[3];
  f.u[4] = p[8]; f.u[5] = p[9]; f.u[6] = p[10]; f.u[7] = p[11];
  return f.v;
}

// ---------------------------------------------------------------------------
// Tensor Data Mover: DMA a 2D bf16 tile [rows x 32 elems] from global memory
// into LDS, padding each 64B row with 8B so the LDS row stride is 18 DWORDs
// (matches load_frag's bank-decorrelated layout).  Descriptor bit layout per
// CDNA5 ISA.  Tracked by TENSORcnt.
// ---------------------------------------------------------------------------
DEV void tdm_load_tile(unsigned int lds_addr, const void* gptr,
                       unsigned int k_elems, unsigned int rows) {
  unsigned long long ga = (unsigned long long)gptr;
  u32x4 g0;
  g0[0] = 1u;                                           // count=1, user mode
  g0[1] = lds_addr;                                     // lds_addr [63:32]
  g0[2] = (unsigned int)ga;                             // global_addr lo
  g0[3] = (unsigned int)((ga >> 32) & 0x01FFFFFFu)      // global_addr [56:32]
          | 0x80000000u;                                // type=2 ("image")
  i32x8 g1;
  g1[0] = (int)((1u << 16)                              // data_size = 2 bytes
          | (1u << 20)                                  // pad_enable
          | (3u << 22)                                  // pad_interval: 16 DW (64B)
          | (1u << 25));                                // pad_amount: 2 DW (8B)
  g1[1] = (int)((k_elems & 0xFFFFu) << 16);             // tensor_dim0 lo16
  g1[2] = (int)(((k_elems >> 16) & 0xFFFFu)             // tensor_dim0 hi16
          | 0xFFFF0000u);                               // tensor_dim1 lo16 (large)
  g1[3] = (int)(32u << 16);                             // dim1 hi=0; tile_dim0=32
  g1[4] = (int)(rows & 0xFFFFu);                        // tile_dim1=rows; tile_dim2=0
  g1[5] = (int)k_elems;                                 // tensor_dim0_stride lo32
  g1[6] = 0;                                            // stride hi; dim1_stride lo
  g1[7] = 0;                                            // dim1_stride hi (unused, 2D)
  i32x4 z4 = {0, 0, 0, 0};                              // groups 2/3: 2D tensor
  i32x8 z8 = {0, 0, 0, 0, 0, 0, 0, 0};                  // unused extra group
  __builtin_amdgcn_tensor_load_to_lds(g0, g1, z4, z4, z8, 0);
}

DEV unsigned int lds_off_u32(const unsigned int* p) {
  // Flat pointers into LDS carry the LDS byte address in their low 32 bits.
  return (unsigned int)(unsigned long long)(const void*)p;
}

// ---------------------------------------------------------------------------
// Utility kernels
// ---------------------------------------------------------------------------
__global__ __launch_bounds__(256) void k_fill_zero(float* p, int n) {
  int i = blockIdx.x * 256 + threadIdx.x;
  if (i < n) p[i] = 0.f;
}

__global__ __launch_bounds__(256) void k_cvt_bf16(const float* in, unsigned short* out, int n) {
  for (int i = blockIdx.x * 256 + threadIdx.x; i < n; i += gridDim.x * 256)
    out[i] = f2bf(in[i]);
}

// Vt[((b*16+h)*64+dh)*2048 + s] = V part of qkv (cols [2048,3072))
__global__ __launch_bounds__(256) void k_transpose_v(const unsigned short* qkv, unsigned short* vt) {
  int i = blockIdx.x * 256 + threadIdx.x;          // over 2*16*64*2048 = 4.19M
  if (i >= 2 * 16 * 64 * 2048) return;
  int s  =  i        & 2047;
  int dh = (i >> 11) & 63;
  int h  = (i >> 17) & 15;
  int b  =  i >> 21;
  vt[i] = qkv[(b * 2048 + s) * 3072 + 2048 + h * 64 + dh];
}

// ---------------------------------------------------------------------------
// bf16 WMMA GEMM:  out[M,N] = A[M,K] @ W[N,K]^T + bias
// 256 threads = 8 waves; block tile 128(M) x 128(N); wave tile 32x64.
// TDM-fed double-buffered LDS: wave 0 issues tensor_load_to_lds for the next
// k-slice while all waves compute the current one.
// Epilogues: 0=f32 store, 1=bf16, 2=relu->bf16, 3=out += gate[row]*(acc+bias)
// ---------------------------------------------------------------------------
enum { EPI_F32 = 0, EPI_BF16 = 1, EPI_RELU_BF16 = 2, EPI_GATE_ACC = 3 };

template <int EPI>
__global__ __launch_bounds__(256) void k_gemm_bf16(
    const unsigned short* __restrict__ A, const unsigned short* __restrict__ W,
    const float* __restrict__ bias, void* __restrict__ out,
    const float* __restrict__ gate, int gstride, int M, int N, int Kd) {
  __shared__ unsigned int ldsA[2][128 * 18];   // padded row stride: 18 u32
  __shared__ unsigned int ldsB[2][128 * 18];

  int tid  = threadIdx.x;
  int lane = tid & 31;
  int wid  = tid >> 5;
  int bm = blockIdx.x * 128, bn = blockIdx.y * 128;
  int wm = (wid >> 1) * 32, wn = (wid & 1) * 64;
  bool issuer = (wid == 0);

  const char* Abase = (const char*)A + (size_t)bm * Kd * 2;
  const char* Wbase = (const char*)W + (size_t)bn * Kd * 2;

  v8f c[2][4] = {};

  int nk = Kd >> 5;
  if (issuer && lane == 0) {
    tdm_load_tile(lds_off_u32(ldsA[0]), Abase, (unsigned)Kd, 128u);
    tdm_load_tile(lds_off_u32(ldsB[0]), Wbase, (unsigned)Kd, 128u);
  }
  for (int ki = 0; ki < nk; ++ki) {
    int buf = ki & 1;
    if (issuer) __builtin_amdgcn_s_wait_tensorcnt(0);
    __syncthreads();                                   // tiles[buf] visible to all
    if (issuer && lane == 0 && ki + 1 < nk) {          // prefetch next slice
      size_t boff = (size_t)(ki + 1) * 64;             // 32 elems * 2B
      tdm_load_tile(lds_off_u32(ldsA[buf ^ 1]), Abase + boff, (unsigned)Kd, 128u);
      tdm_load_tile(lds_off_u32(ldsB[buf ^ 1]), Wbase + boff, (unsigned)Kd, 128u);
    }
    const unsigned int* la = ldsA[buf];
    const unsigned int* lb = ldsB[buf];
    v16bf a0 = load_frag(la, wm + 0,  18, 0);
    v16bf a1 = load_frag(la, wm + 16, 18, 0);
    v16bf b0 = load_frag(lb, wn + 0,  18, 0);
    v16bf b1 = load_frag(lb, wn + 16, 18, 0);
    v16bf b2 = load_frag(lb, wn + 32, 18, 0);
    v16bf b3 = load_frag(lb, wn + 48, 18, 0);

    c[0][0] = wmma_bf16(a0, b0, c[0][0]);
    c[0][1] = wmma_bf16(a0, b1, c[0][1]);
    c[0][2] = wmma_bf16(a0, b2, c[0][2]);
    c[0][3] = wmma_bf16(a0, b3, c[0][3]);
    c[1][0] = wmma_bf16(a1, b0, c[1][0]);
    c[1][1] = wmma_bf16(a1, b1, c[1][1]);
    c[1][2] = wmma_bf16(a1, b2, c[1][2]);
    c[1][3] = wmma_bf16(a1, b3, c[1][3]);
    __syncthreads();                                   // done reading tiles[buf]
  }

  // Epilogue.  C/D layout: VGPR j -> row 8*(lane>>4)+j, col = lane&15.
  int half = lane >> 4, nlo = lane & 15;
#pragma unroll
  for (int mt = 0; mt < 2; ++mt) {
#pragma unroll
    for (int nt = 0; nt < 4; ++nt) {
      int col = bn + wn + nt * 16 + nlo;
      float bv = bias ? bias[col] : 0.f;
#pragma unroll
      for (int j = 0; j < 8; ++j) {
        int row = bm + wm + mt * 16 + half * 8 + j;
        float v = c[mt][nt][j] + bv;
        int idx = row * N + col;
        if (EPI == EPI_F32) {
          ((float*)out)[idx] = v;
        } else if (EPI == EPI_BF16) {
          ((unsigned short*)out)[idx] = f2bf(v);
        } else if (EPI == EPI_RELU_BF16) {
          ((unsigned short*)out)[idx] = f2bf(fmaxf(v, 0.f));
        } else {  // EPI_GATE_ACC
          float g = gate[row * gstride];
          ((float*)out)[idx] += g * v;     // per-element unique -> no race
        }
      }
    }
  }
}

// ---------------------------------------------------------------------------
// Attention: 1 block = (b, h, 64 q rows); 4 waves, each wave = 16 q rows.
// Two-pass exact softmax; emits head-averaged pattern via atomic f32 adds.
// qkv: bf16 [4096, 3072] (q|k|v).  Vt: bf16 [(b*16+h)*64+dh][2048].
// ---------------------------------------------------------------------------
__global__ __launch_bounds__(128) void k_attention(
    const unsigned short* __restrict__ qkv_, const unsigned short* __restrict__ vt_,
    unsigned short* __restrict__ o_, float* __restrict__ pattern) {
  __shared__ unsigned int ldsP[4][16 * 18];     // per-wave P tile (16x32 bf16)

  const unsigned int* qkv = (const unsigned int*)qkv_;  // row stride 1536 u32
  const unsigned int* vt  = (const unsigned int*)vt_;   // row stride 1024 u32

  int lane = threadIdx.x & 31;
  int wid  = threadIdx.x >> 5;
  int qblk =  blockIdx.x        & 31;
  int h    = (blockIdx.x >> 5)  & 15;
  int b    =  blockIdx.x >> 9;
  int tok0 = b * 2048;
  int q0   = qblk * 64 + wid * 16;
  int half = lane >> 4, nlo = lane & 15;
  const float scale = 0.125f;                   // 1/sqrt(64)

  // Q fragments live in registers across both passes (K dim = 64 -> 2 frags)
  v16bf qa0 = load_frag(qkv, tok0 + q0, 1536, h * 32);
  v16bf qa1 = load_frag(qkv, tok0 + q0, 1536, h * 32 + 16);

  float mrow[8], lrow[8];
#pragma unroll
  for (int j = 0; j < 8; ++j) { mrow[j] = -3.0e38f; lrow[j] = 0.f; }

  // ---- pass 1: exact row max & denominator -------------------------------
  for (int kc = 0; kc < 2048; kc += 16) {
    v16bf kb0 = load_frag(qkv, tok0 + kc, 1536, 512 + h * 32);
    v16bf kb1 = load_frag(qkv, tok0 + kc, 1536, 512 + h * 32 + 16);
    v8f s = {};
    s = wmma_bf16(qa0, kb0, s);
    s = wmma_bf16(qa1, kb1, s);
#pragma unroll
    for (int j = 0; j < 8; ++j) {
      float sv = s[j] * scale;
      float rmax = sv;
      for (int o = 1; o < 16; o <<= 1) rmax = fmaxf(rmax, __shfl_xor(rmax, o, 16));
      float nm = fmaxf(mrow[j], rmax);
      float ex = __expf(sv - nm);
      for (int o = 1; o < 16; o <<= 1) ex += __shfl_xor(ex, o, 16);
      lrow[j] = lrow[j] * __expf(mrow[j] - nm) + ex;
      mrow[j] = nm;
    }
  }
  float inv[8];
#pragma unroll
  for (int j = 0; j < 8; ++j) inv[j] = 1.f / lrow[j];

  // ---- pass 2: normalized P, pattern atomics, O = P @ V ------------------
  v8f co0 = {}, co1 = {}, co2 = {}, co3 = {};
  unsigned short* lp = (unsigned short*)&ldsP[wid][0];  // row stride 36 bf16
  int vrow0 = (b * 16 + h) * 64;

  for (int kc = 0; kc < 2048; kc += 32) {
#pragma unroll
    for (int s2 = 0; s2 < 2; ++s2) {
      int kb = kc + s2 * 16;
      v16bf kb0 = load_frag(qkv, tok0 + kb, 1536, 512 + h * 32);
      v16bf kb1 = load_frag(qkv, tok0 + kb, 1536, 512 + h * 32 + 16);
      v8f s = {};
      s = wmma_bf16(qa0, kb0, s);
      s = wmma_bf16(qa1, kb1, s);
#pragma unroll
      for (int j = 0; j < 8; ++j) {
        float p = __expf(s[j] * scale - mrow[j]) * inv[j];
        int row = half * 8 + j;
        atomicAdd(pattern + (tok0 + q0 + row) * 2048 + kb + nlo, p * 0.0625f);
        lp[row * 36 + s2 * 16 + nlo] = f2bf(p);   // stage P for A-frag reload
      }
    }
    // P tile (16 q x 32 k) -> A fragment; V chunk (32 k x 64 dh) -> 4 B frags
    v16bf pa = load_frag(ldsP[wid], 0, 18, 0);
    int cu = kc >> 1;
    co0 = wmma_bf16(pa, load_frag(vt, vrow0 +  0, 1024, cu), co0);
    co1 = wmma_bf16(pa, load_frag(vt, vrow0 + 16, 1024, cu), co1);
    co2 = wmma_bf16(pa, load_frag(vt, vrow0 + 32, 1024, cu), co2);
    co3 = wmma_bf16(pa, load_frag(vt, vrow0 + 48, 1024, cu), co3);
  }

  v8f cos[4] = {co0, co1, co2, co3};
#pragma unroll
  for (int nt = 0; nt < 4; ++nt) {
    int col = h * 64 + nt * 16 + nlo;
#pragma unroll
    for (int j = 0; j < 8; ++j) {
      int row = tok0 + q0 + half * 8 + j;
      o_[row * 1024 + col] = f2bf(cos[nt][j]);
    }
  }
}

// ---------------------------------------------------------------------------
// LayerNorm over D=1024: out = LN(a + b) * g + beta ; optional bf16 copy.
// ---------------------------------------------------------------------------
__global__ __launch_bounds__(256) void k_layernorm(
    const float* __restrict__ a, const float* __restrict__ bsum,
    const float* __restrict__ g, const float* __restrict__ beta,
    float* __restrict__ out32, unsigned short* __restrict__ out16) {
  __shared__ float red[256];
  int t = blockIdx.x;
  const float* ar = a + t * 1024;
  const float* br = bsum + t * 1024;
  float v[4], s0 = 0.f, s1 = 0.f;
#pragma unroll
  for (int i = 0; i < 4; ++i) {
    int d = threadIdx.x + i * 256;
    float x = ar[d] + br[d];
    v[i] = x; s0 += x; s1 += x * x;
  }
  red[threadIdx.x] = s0; __syncthreads();
  for (int o = 128; o > 0; o >>= 1) { if (threadIdx.x < o) red[threadIdx.x] += red[threadIdx.x + o]; __syncthreads(); }
  float mean = red[0] * (1.f / 1024.f); __syncthreads();
  red[threadIdx.x] = s1; __syncthreads();
  for (int o = 128; o > 0; o >>= 1) { if (threadIdx.x < o) red[threadIdx.x] += red[threadIdx.x + o]; __syncthreads(); }
  float var = red[0] * (1.f / 1024.f) - mean * mean;
  float rinv = rsqrtf(var + 1e-5f);
#pragma unroll
  for (int i = 0; i < 4; ++i) {
    int d = threadIdx.x + i * 256;
    float y = (v[i] - mean) * rinv * g[d] + beta[d];
    if (out32) out32[t * 1024 + d] = y;
    if (out16) out16[t * 1024 + d] = f2bf(y);
  }
}

// ---------------------------------------------------------------------------
// Router: logits = x @ router_w^T; exact top-2 softmax -> dense gates [tok,8].
// ---------------------------------------------------------------------------
__global__ __launch_bounds__(256) void k_router(
    const float* __restrict__ x, const float* __restrict__ rw, float* __restrict__ gates) {
  int wid = threadIdx.x >> 5, lane = threadIdx.x & 31;
  int t = blockIdx.x * 8 + wid;
  const float* xr = x + t * 1024;
  float logit[8];
#pragma unroll
  for (int e = 0; e < 8; ++e) {
    const float* wr = rw + e * 1024;
    float acc = 0.f;
    for (int d = lane; d < 1024; d += 32) acc += xr[d] * wr[d];
    for (int o = 16; o > 0; o >>= 1) acc += __shfl_xor(acc, o, 32);
    logit[e] = acc;
  }
  int i1 = 0; float m1 = logit[0];
#pragma unroll
  for (int e = 1; e < 8; ++e) if (logit[e] > m1) { m1 = logit[e]; i1 = e; }   // first-index tie-break (jax)
  int i2 = -1; float m2 = -3.0e38f;
#pragma unroll
  for (int e = 0; e < 8; ++e) if (e != i1 && logit[e] > m2) { m2 = logit[e]; i2 = e; }
  float e2 = __expf(m2 - m1);
  float p1 = 1.f / (1.f + e2), p2 = e2 / (1.f + e2);
  if (lane < 8) gates[t * 8 + lane] = (lane == i1) ? p1 : ((lane == i2) ? p2 : 0.f);
}

// ---------------------------------------------------------------------------
// Host-side orchestration
// ---------------------------------------------------------------------------
static void launch_gemm(int epi, const unsigned short* A, const unsigned short* W,
                        const float* bias, void* out, const float* gate, int gstride,
                        int M, int N, int Kd, hipStream_t s) {
  dim3 g(M / 128, N / 128), b(256);
  switch (epi) {
    case EPI_F32:       k_gemm_bf16<EPI_F32><<<g, b, 0, s>>>(A, W, bias, out, gate, gstride, M, N, Kd); break;
    case EPI_BF16:      k_gemm_bf16<EPI_BF16><<<g, b, 0, s>>>(A, W, bias, out, gate, gstride, M, N, Kd); break;
    case EPI_RELU_BF16: k_gemm_bf16<EPI_RELU_BF16><<<g, b, 0, s>>>(A, W, bias, out, gate, gstride, M, N, Kd); break;
    default:            k_gemm_bf16<EPI_GATE_ACC><<<g, b, 0, s>>>(A, W, bias, out, gate, gstride, M, N, Kd); break;
  }
}

extern "C" void kernel_launch(void* const* d_in, const int* in_sizes, int n_in,
                              void* d_out, int out_size, void* d_ws, size_t ws_size,
                              hipStream_t stream) {
  (void)in_sizes; (void)n_in; (void)out_size; (void)ws_size;

  const float* src        = (const float*)d_in[0];
  const float* in_proj_w  = (const float*)d_in[1];
  const float* in_proj_b  = (const float*)d_in[2];
  const float* out_proj_w = (const float*)d_in[3];
  const float* out_proj_b = (const float*)d_in[4];
  const float* router_w   = (const float*)d_in[5];
  const float* up_w       = (const float*)d_in[6];
  const float* up_b       = (const float*)d_in[7];
  const float* down_w     = (const float*)d_in[8];
  const float* down_b     = (const float*)d_in[9];
  const float* g1         = (const float*)d_in[10];
  const float* b1         = (const float*)d_in[11];
  const float* g2         = (const float*)d_in[12];
  const float* b2         = (const float*)d_in[13];

  constexpr int TOK = 4096, D = 1024, DFF = 4096;
  float* x_out   = (float*)d_out;                 // [B,S,D]  = 4194304
  float* pattern = (float*)d_out + TOK * D;       // [B,S,S]  = 8388608

  size_t off = 0;
  auto alloc = [&](size_t n) -> void* {
    off = (off + 255) & ~(size_t)255;
    void* p = (char*)d_ws + off; off += n; return p;
  };
  unsigned short* src16   = (unsigned short*)alloc((size_t)TOK * D * 2);
  unsigned short* wqkv16  = (unsigned short*)alloc((size_t)3 * D * D * 2);
  unsigned short* wout16  = (unsigned short*)alloc((size_t)D * D * 2);
  unsigned short* qkv16   = (unsigned short*)alloc((size_t)TOK * 3 * D * 2);
  unsigned short* vt16    = (unsigned short*)alloc((size_t)TOK * D * 2);
  unsigned short* o16     = (unsigned short*)alloc((size_t)TOK * D * 2);
  float*          attnres = (float*)alloc((size_t)TOK * D * 4);
  float*          x1f     = (float*)alloc((size_t)TOK * D * 4);
  unsigned short* x1h     = (unsigned short*)alloc((size_t)TOK * D * 2);
  float*          gates   = (float*)alloc((size_t)TOK * 8 * 4);
  unsigned short* upe16   = (unsigned short*)alloc((size_t)DFF * D * 2);
  unsigned short* dne16   = (unsigned short*)alloc((size_t)D * DFF * 2);
  unsigned short* h16     = (unsigned short*)alloc((size_t)TOK * DFF * 2);
  float*          moe     = (float*)alloc((size_t)TOK * D * 4);

  // Zero-init buffers that are accumulated into (harness poisons, never re-zeroes).
  k_fill_zero<<<(TOK * 2048 + 255) / 256, 256, 0, stream>>>(pattern, TOK * 2048);
  k_fill_zero<<<(TOK * D + 255) / 256, 256, 0, stream>>>(moe, TOK * D);

  // f32 -> bf16 conversions
  k_cvt_bf16<<<(TOK * D + 255) / 256, 256, 0, stream>>>(src, src16, TOK * D);
  k_cvt_bf16<<<(3 * D * D + 255) / 256, 256, 0, stream>>>(in_proj_w, wqkv16, 3 * D * D);
  k_cvt_bf16<<<(D * D + 255) / 256, 256, 0, stream>>>(out_proj_w, wout16, D * D);

  // QKV projection (bf16 out), V transpose, attention, out projection (f32 out)
  launch_gemm(EPI_BF16, src16, wqkv16, in_proj_b, qkv16, nullptr, 0, TOK, 3 * D, D, stream);
  k_transpose_v<<<(TOK * D + 255) / 256, 256, 0, stream>>>(qkv16, vt16);
  k_attention<<<1024, 128, 0, stream>>>(qkv16, vt16, o16, pattern);
  launch_gemm(EPI_F32, o16, wout16, out_proj_b, attnres, nullptr, 0, TOK, D, D, stream);

  // x1 = LN1(src + attn) ; router gates
  k_layernorm<<<TOK, 256, 0, stream>>>(src, attnres, g1, b1, x1f, x1h);
  k_router<<<TOK / 8, 256, 0, stream>>>(x1f, router_w, gates);

  // Experts: h = relu(x1 @ up^T + up_b); moe += gate_e * (h @ down^T + down_b)
  for (int e = 0; e < 8; ++e) {
    k_cvt_bf16<<<(DFF * D + 255) / 256, 256, 0, stream>>>(up_w + (size_t)e * DFF * D, upe16, DFF * D);
    k_cvt_bf16<<<(D * DFF + 255) / 256, 256, 0, stream>>>(down_w + (size_t)e * D * DFF, dne16, D * DFF);
    launch_gemm(EPI_RELU_BF16, x1h, upe16, up_b + e * DFF, h16, nullptr, 0, TOK, DFF, D, stream);
    launch_gemm(EPI_GATE_ACC, h16, dne16, down_b + e * D, moe, gates + e, 8, TOK, D, DFF, stream);
  }

  // x = LN2(x1 + moe)
  k_layernorm<<<TOK, 256, 0, stream>>>(x1f, moe, g2, b2, x_out, nullptr);
}
